// KdeDiscriminator_51900384805285
// MI455X (gfx1250) — compile-verified, matrix operand order checked by reading, not run on hardware
//
#include <hip/hip_runtime.h>
#include <hip/hip_bf16.h>

typedef __attribute__((ext_vector_type(16))) _Float16 v16h;
typedef __attribute__((ext_vector_type(8)))  float    v8f;

#define D_DIM 64
#define WAVES_PER_BLOCK 8

// Build one lane's 16-element f16 fragment (CDNA5 16-bit WMMA operand layout)
// from a row-major fp32 row of 64 elements. kh selects K in [0,32) or [32,64).
// Lanes 0-15 carry K sub-runs {base..base+7, base+16..base+23} with
// base = kh*32; lanes 16-31 carry the +8 shifted runs.
__device__ __forceinline__ v16h load_frag_f16(const float* __restrict__ rp,
                                              int kh, int khalf) {
  const float4* p = (const float4*)(rp + kh * 32 + khalf * 8);
  float4 q0 = p[0], q1 = p[1];   // K = kh*32 + khalf*8 .. +7
  float4 q2 = p[4], q3 = p[5];   // K = kh*32 + khalf*8 + 16 .. +23
  float buf[16] = {q0.x, q0.y, q0.z, q0.w, q1.x, q1.y, q1.z, q1.w,
                   q2.x, q2.y, q2.z, q2.w, q3.x, q3.y, q3.z, q3.w};
  v16h f;
#pragma unroll
  for (int i = 0; i < 16; ++i) f[i] = (_Float16)buf[i];
  return f;
}

// Kernel 1: exact fp32 row norms ||row||^2 for a [rows, 64] matrix.
__global__ void kde_row_norms(const float* __restrict__ a,
                              float* __restrict__ out, int rows) {
  int r = blockIdx.x * blockDim.x + threadIdx.x;
  if (r >= rows) return;
  const float4* p = (const float4*)(a + (size_t)r * D_DIM);
  float s = 0.f;
#pragma unroll
  for (int i = 0; i < D_DIM / 4; ++i) {
    float4 v = p[i];
    s = fmaf(v.x, v.x, s);
    s = fmaf(v.y, v.y, s);
    s = fmaf(v.z, v.z, s);
    s = fmaf(v.w, v.w, s);
  }
  out[r] = s;
}

// Kernel 2: one-shot fp32 -> f16 conversion, pre-swizzled into per-lane WMMA
// B-fragment layout. Entry t = tile*64 + kh*32 + lane holds the 16 f16 values
// lane `lane` feeds to WMMA for K-half `kh` of 16-row tile `tile`.
__global__ void kde_pack_frags(const float* __restrict__ a,
                               v16h* __restrict__ out, int mtiles) {
  int t = blockIdx.x * blockDim.x + threadIdx.x;
  if (t >= mtiles * 64) return;
  int lane  = t & 31;
  int kh    = (t >> 5) & 1;
  int tile  = t >> 6;
  int row   = tile * 16 + (lane & 15);
  int khalf = lane >> 4;
  out[t] = load_frag_f16(a + (size_t)row * D_DIM, kh, khalf);
}

// Kernel 3: fused distance-GEMM + exp + row-sum + ratio.
// One block per 16-query tile; 8 waves split the M dimension.
__global__ __launch_bounds__(256, 1) void kde_disc_kernel(
    const float* __restrict__ x,
    const v16h* __restrict__ pack0, const v16h* __restrict__ pack1,
    const float* __restrict__ nq,
    const float* __restrict__ n0, const float* __restrict__ n1,
    float* __restrict__ out, int M) {
  __shared__ float red0[WAVES_PER_BLOCK][16];
  __shared__ float red1[WAVES_PER_BLOCK][16];

  const int tid   = threadIdx.x;
  const int wave  = tid >> 5;
  const int lane  = tid & 31;
  const int l16   = lane & 15;
  const int khalf = lane >> 4;
  const int qbase = blockIdx.x * 16;

  // Resident A fragments: query rows qbase..qbase+15, K = 0..63 (fp32 -> f16
  // converted once per block).
  const float* arow = x + (size_t)(qbase + l16) * D_DIM;
  v16h aLo = load_frag_f16(arow, 0, khalf);
  v16h aHi = load_frag_f16(arow, 1, khalf);

  // fp32 query norms for this lane's 8 C-matrix row slots
  // (VGPR v holds row M=v for lanes 0-15, M=v+8 for lanes 16-31).
  const int rowoff = khalf * 8;
  float qq[8];
#pragma unroll
  for (int v = 0; v < 8; ++v) qq[v] = nq[qbase + rowoff + v];

  float s0[8], s1[8];
#pragma unroll
  for (int v = 0; v < 8; ++v) { s0[v] = 0.f; s1[v] = 0.f; }

  const int tilesTotal   = M / 16;
  const int tilesPerWave = tilesTotal / WAVES_PER_BLOCK;
  const int tile0        = wave * tilesPerWave;
  const float kexp       = -12.5f;  // -1 / (2 * 0.2 * 0.2)

  for (int t = 0; t < tilesPerWave; ++t) {
    const int tile = tile0 + t;
    const size_t fbase = (size_t)tile * 64 + lane;

    if (t + 1 < tilesPerWave) {   // stream next tile's fragments into cache
      __builtin_prefetch(&pack0[fbase + 64], 0, 1);
      __builtin_prefetch(&pack1[fbase + 64], 0, 1);
    }

    // Pre-swizzled f16 B fragments: pure b128 loads, no conversion VALU.
    v16h b0lo = pack0[fbase];
    v16h b0hi = pack0[fbase + 32];
    v16h b1lo = pack1[fbase];
    v16h b1hi = pack1[fbase + 32];

    // Two independent accumulator chains (x0 / x1) for WMMA ILP.
    v8f c0 = {}, c1 = {};
    c0 = __builtin_amdgcn_wmma_f32_16x16x32_f16(false, aLo, false, b0lo,
                                                (short)0, c0, false, false);
    c1 = __builtin_amdgcn_wmma_f32_16x16x32_f16(false, aLo, false, b1lo,
                                                (short)0, c1, false, false);
    c0 = __builtin_amdgcn_wmma_f32_16x16x32_f16(false, aHi, false, b0hi,
                                                (short)0, c0, false, false);
    c1 = __builtin_amdgcn_wmma_f32_16x16x32_f16(false, aHi, false, b1hi,
                                                (short)0, c1, false, false);

    const int nb = tile * 16;
    const float dd0 = n0[nb + l16];   // column norm (col N = lane%16)
    const float dd1 = n1[nb + l16];
#pragma unroll
    for (int v = 0; v < 8; ++v) {
      float sq0 = fmaxf(qq[v] + dd0 - 2.f * c0[v], 0.f);
      float sq1 = fmaxf(qq[v] + dd1 - 2.f * c1[v], 0.f);
      s0[v] += __expf(kexp * sq0);    // v_exp_f32, co-executes with WMMA
      s1[v] += __expf(kexp * sq1);
    }
  }

  // Reduce over the 16 columns held across each 16-lane half (wave32 shuffles).
#pragma unroll
  for (int m = 1; m <= 8; m <<= 1) {
#pragma unroll
    for (int v = 0; v < 8; ++v) {
      s0[v] += __shfl_xor(s0[v], m, 32);
      s1[v] += __shfl_xor(s1[v], m, 32);
    }
  }
  if (l16 == 0) {
#pragma unroll
    for (int v = 0; v < 8; ++v) {
      red0[wave][rowoff + v] = s0[v];
      red1[wave][rowoff + v] = s1[v];
    }
  }
  __syncthreads();

  // Cross-wave reduction + final ratio (norm constant and 1/M cancel).
  if (tid < 16) {
    float S0 = 0.f, S1 = 0.f;
#pragma unroll
    for (int w = 0; w < WAVES_PER_BLOCK; ++w) {
      S0 += red0[w][tid];
      S1 += red1[w][tid];
    }
    out[qbase + tid] = S1 / (S0 + S1);
  }
}

extern "C" void kernel_launch(void* const* d_in, const int* in_sizes, int n_in,
                              void* d_out, int out_size, void* d_ws,
                              size_t ws_size, hipStream_t stream) {
  const float* x  = (const float*)d_in[0];
  const float* x0 = (const float*)d_in[1];
  const float* x1 = (const float*)d_in[2];
  float* out = (float*)d_out;

  const int N = in_sizes[0] / D_DIM;   // 8192
  const int M = in_sizes[1] / D_DIM;   // 8192
  const int mtiles = M / 16;

  // Workspace layout: fp32 norms, then pre-packed f16 fragments (32B aligned:
  // (N+2M)*4 bytes is a multiple of 32).
  float* nq = (float*)d_ws;            // N fp32 query norms
  float* n0 = nq + N;                  // M fp32 class-0 norms
  float* n1 = n0 + M;                  // M fp32 class-1 norms
  v16h* pack0 = (v16h*)(n1 + M);       // mtiles*64 fragments for x0
  v16h* pack1 = pack0 + (size_t)mtiles * 64;

  kde_row_norms<<<(N + 255) / 256, 256, 0, stream>>>(x,  nq, N);
  kde_row_norms<<<(M + 255) / 256, 256, 0, stream>>>(x0, n0, M);
  kde_row_norms<<<(M + 255) / 256, 256, 0, stream>>>(x1, n1, M);

  const int packThreads = mtiles * 64;
  kde_pack_frags<<<(packThreads + 255) / 256, 256, 0, stream>>>(x0, pack0, mtiles);
  kde_pack_frags<<<(packThreads + 255) / 256, 256, 0, stream>>>(x1, pack1, mtiles);

  kde_disc_kernel<<<N / 16, 256, 0, stream>>>(x, pack0, pack1, nq, n0, n1,
                                              out, M);
}
